// MaskedMultiHeadSelfAttention_65403761983824
// MI455X (gfx1250) — matrix operand order repaired and checked by reading, not compiled, not run on hardware
//
#include <hip/hip_runtime.h>

typedef __attribute__((ext_vector_type(16))) _Float16     v16h;
typedef __attribute__((ext_vector_type(8)))  float        v8f;
typedef __attribute__((ext_vector_type(4)))  unsigned int u32x4;
typedef __attribute__((ext_vector_type(4)))  int          i32x4;
typedef __attribute__((ext_vector_type(4)))  float        f32x4;
typedef __attribute__((ext_vector_type(4)))  unsigned short u16x4;

constexpr int kEmbed = 768;
constexpr int kSeq   = 1024;
constexpr int kB     = 8;
constexpr int kH     = 12;
constexpr int kHD    = 64;
constexpr int LDT    = 72;   // padded LDS row: 144 B = 9*16 B -> b128-aligned rows

// ---- CDNA5 async global->LDS copy (ASYNCcnt path), with safe fallback ------
#if defined(__has_builtin)
#if __has_builtin(__builtin_amdgcn_global_load_async_to_lds_b128)
#define HAVE_ASYNC_LDS 1
#endif
#endif
#ifndef HAVE_ASYNC_LDS
#define HAVE_ASYNC_LDS 0
#endif

#if HAVE_ASYNC_LDS
static __device__ __forceinline__ void async_copy16(const unsigned short* g,
                                                    unsigned short* l) {
  // toolchain signature (from diagnostic): (int4* gsrc, int4* ldst, Imm, Imm)
  __builtin_amdgcn_global_load_async_to_lds_b128(
      (i32x4*)const_cast<unsigned short*>(g), (i32x4*)l,
      /*offset=*/0, /*cpol=*/0);
}
static __device__ __forceinline__ void async_wait0() {
#if __has_builtin(__builtin_amdgcn_s_wait_asynccnt)
  __builtin_amdgcn_s_wait_asynccnt(0);
#else
  asm volatile("s_wait_asynccnt 0x0" ::: "memory");
#endif
}
#endif

static __device__ __forceinline__ unsigned short f2h(float f) {
  _Float16 h = (_Float16)f;
  return __builtin_bit_cast(unsigned short, h);
}

union FragU { u32x4 q[2]; v16h v; unsigned short s[16]; };

// A-operand (16x32 f16, M rows striped per lane): lanes0-15 K = k0+{0..7} and
// k0+16+{0..7}; lanes16-31 same +8 (ISA 7.12.2, 16-bit A 16x32).
static __device__ __forceinline__ v16h ld_frag_a(const unsigned short* base, int ld,
                                                 int lane, int k0) {
  const int m = lane & 15, sel = lane >> 4;
  const unsigned short* p = base + (size_t)m * ld + k0 + sel * 8;
  FragU f;
  f.q[0] = *reinterpret_cast<const u32x4*>(p);
  f.q[1] = *reinterpret_cast<const u32x4*>(p + 16);
  return f.v;
}

// B-operand (32x16 f16) loaded from row-major B^T[n][k]: lane n holds column n,
// lanes0-15 K = k0+{0..7} and k0+16+{0..7}; lanes16-31 same +8.
static __device__ __forceinline__ v16h ld_frag_b(const unsigned short* base, int ld,
                                                 int lane, int k0) {
  const int n = lane & 15, sel = lane >> 4;
  const unsigned short* p = base + (size_t)n * ld + k0 + sel * 8;
  FragU f;
  f.q[0] = *reinterpret_cast<const u32x4*>(p);
  f.q[1] = *reinterpret_cast<const u32x4*>(p + 16);
  return f.v;
}

static __device__ __forceinline__ v8f wmma_f16(v16h a, v16h b, v8f c) {
  return __builtin_amdgcn_wmma_f32_16x16x32_f16(false, a, false, b, (short)0, c,
                                                false, false);
}

__global__ void cvt_f32_f16(const float* __restrict__ src,
                            unsigned short* __restrict__ dst, int n4) {
  int i = blockIdx.x * blockDim.x + threadIdx.x;
  if (i < n4) {
    f32x4 v = reinterpret_cast<const f32x4*>(src)[i];
    u16x4 o;
    o[0] = f2h(v[0]); o[1] = f2h(v[1]); o[2] = f2h(v[2]); o[3] = f2h(v[3]);
    reinterpret_cast<u16x4*>(dst)[i] = o;
  }
}

// C[M,N] = A[M,K] @ Bw[N,K]^T + bias ; A/Bw f16 row-major, out f16 or f32.
// Block tile 128x128, BK=64, 256 threads = 8 waves (2x4), wave tile 64x32.
__global__ void __launch_bounds__(256)
gemm_nt_f16(const unsigned short* __restrict__ A,
            const unsigned short* __restrict__ Bw,
            const float* __restrict__ bias, void* __restrict__ Cout,
            int M, int N, int K, int out_f32) {
  __shared__ unsigned short As[128 * LDT];
  __shared__ unsigned short Bs[128 * LDT];
  const int tid = threadIdx.x, lane = tid & 31, wid = tid >> 5;
  const int wm = wid >> 2, wn = wid & 3;
  const int m0 = blockIdx.y * 128, n0 = blockIdx.x * 128;
  const int sel = lane >> 4, nlo = lane & 15;

  v8f acc[4][2];
  for (int ni = 0; ni < 2; ++ni) {
    float bv = bias[n0 + wn * 32 + ni * 16 + nlo];
    for (int mi = 0; mi < 4; ++mi)
      for (int e = 0; e < 8; ++e) acc[mi][ni][e] = bv;
  }

  // this thread's two 16B chunks of the 128x64 tile (per matrix)
  const int row0 = tid >> 2,        c0 = (tid & 3) << 4;
  const int row1 = (tid + 256) >> 2, c1 = ((tid + 256) & 3) << 4;

  const int kblocks = K >> 6;
  for (int kb = 0; kb < kblocks; ++kb) {
    const int k0 = kb << 6;
#if HAVE_ASYNC_LDS
    async_copy16(&A[(size_t)(m0 + row0) * K + k0 + c0], &As[row0 * LDT + c0]);
    async_copy16(&A[(size_t)(m0 + row1) * K + k0 + c1], &As[row1 * LDT + c1]);
    async_copy16(&Bw[(size_t)(n0 + row0) * K + k0 + c0], &Bs[row0 * LDT + c0]);
    async_copy16(&Bw[(size_t)(n0 + row1) * K + k0 + c1], &Bs[row1 * LDT + c1]);
    async_wait0();
#else
    *reinterpret_cast<u32x4*>(&As[row0 * LDT + c0]) =
        *reinterpret_cast<const u32x4*>(&A[(size_t)(m0 + row0) * K + k0 + c0]);
    *reinterpret_cast<u32x4*>(&As[row1 * LDT + c1]) =
        *reinterpret_cast<const u32x4*>(&A[(size_t)(m0 + row1) * K + k0 + c1]);
    *reinterpret_cast<u32x4*>(&Bs[row0 * LDT + c0]) =
        *reinterpret_cast<const u32x4*>(&Bw[(size_t)(n0 + row0) * K + k0 + c0]);
    *reinterpret_cast<u32x4*>(&Bs[row1 * LDT + c1]) =
        *reinterpret_cast<const u32x4*>(&Bw[(size_t)(n0 + row1) * K + k0 + c1]);
#endif
    if (kb + 1 < kblocks) {  // warm L2/WGP$ for next k-slab
      __builtin_prefetch(&A[(size_t)(m0 + row0) * K + k0 + 64 + c0], 0, 0);
      __builtin_prefetch(&Bw[(size_t)(n0 + row0) * K + k0 + 64 + c0], 0, 0);
    }
    __syncthreads();
    for (int ks = 0; ks < 2; ++ks) {
      v16h af[4], bf[2];
      for (int mi = 0; mi < 4; ++mi)
        af[mi] = ld_frag_a(&As[(wm * 64 + mi * 16) * LDT], LDT, lane, ks * 32);
      for (int ni = 0; ni < 2; ++ni)
        bf[ni] = ld_frag_b(&Bs[(wn * 32 + ni * 16) * LDT], LDT, lane, ks * 32);
      for (int mi = 0; mi < 4; ++mi)
        for (int ni = 0; ni < 2; ++ni)
          acc[mi][ni] = wmma_f16(af[mi], bf[ni], acc[mi][ni]);
    }
    __syncthreads();
  }

  for (int mi = 0; mi < 4; ++mi)
    for (int ni = 0; ni < 2; ++ni) {
      int col = n0 + wn * 32 + ni * 16 + nlo;
      for (int r = 0; r < 8; ++r) {  // C-layout: VGPR r -> row r (+8 for hi lanes)
        int row = m0 + wm * 64 + mi * 16 + sel * 8 + r;
        float v = acc[mi][ni][r];
        if (out_f32)
          reinterpret_cast<float*>(Cout)[(size_t)row * N + col] = v;
        else
          reinterpret_cast<unsigned short*>(Cout)[(size_t)row * N + col] = f2h(v);
      }
    }
}

// Flash attention over K==Q==V. One workgroup (4 waves) = 64 query rows of one
// (b,h); streams 64-key tiles with online softmax; fp32 accumulation.
__global__ void __launch_bounds__(128)
flash_attn_kernel(const unsigned short* __restrict__ Kh,  // [B,L,768] f16
                  const int* __restrict__ amask,          // [B,L]
                  unsigned short* __restrict__ wV) {      // [B,L,768] f16
  __shared__ unsigned short Kn[64 * LDT];   // Kj row-major (j x d)
  __shared__ unsigned short Vt[64 * LDT];   // Kj transposed (d x j) for P@V B-operand
  __shared__ unsigned short Pb[4][16 * LDT];// per-wave P staging (C-layout -> A-layout)

  const int tid = threadIdx.x, lane = tid & 31, wid = tid >> 5;
  const int sel = lane >> 4, nlo = lane & 15;
  const int i0 = blockIdx.x * 64;
  const int b = blockIdx.y / kH, h = blockIdx.y % kH;
  const float sm_scale = 0.03608439182435161f;  // 1/sqrt(768)

  // Ki A-fragments (this wave's 16 rows, d=0..63) live in registers all kernel
  const unsigned short* KiBase =
      Kh + (size_t)(b * kSeq + i0 + wid * 16) * kEmbed + h * kHD;
  v16h aKi[2];
  aKi[0] = ld_frag_a(KiBase, kEmbed, lane, 0);
  aKi[1] = ld_frag_a(KiBase, kEmbed, lane, 32);

  float m_st[8], l_st[8], keep[8];
  for (int r = 0; r < 8; ++r) {
    m_st[r] = -3.0e38f; l_st[r] = 0.f;
    int row = i0 + wid * 16 + sel * 8 + r;
    keep[r] = (amask[b * kSeq + row] != 0) ? 1.f : 0.f;
  }
  v8f o_acc[4];
  for (int t = 0; t < 4; ++t)
    for (int e = 0; e < 8; ++e) o_acc[t][e] = 0.f;

  for (int jt = 0; jt < kSeq / 64; ++jt) {
    const int j0 = jt * 64;
    for (int p = 0; p < 2; ++p) {   // load Kj 64x64 + build transpose
      int idx = tid + (p << 7);
      int row = idx >> 2, c = (idx & 3) << 4;
      const unsigned short* gsrc =
          &Kh[(size_t)(b * kSeq + j0 + row) * kEmbed + h * kHD + c];
      FragU f;
      u32x4 g = *reinterpret_cast<const u32x4*>(gsrc);
      *reinterpret_cast<u32x4*>(&Kn[row * LDT + c]) = g;
      f.q[0] = g;
      for (int e = 0; e < 8; ++e) Vt[(c + e) * LDT + row] = f.s[e];
      if (jt + 1 < kSeq / 64)  // prefetch next key tile (global_prefetch_b8)
        __builtin_prefetch(gsrc + (size_t)64 * kEmbed, 0, 0);
    }
    __syncthreads();

    // S = Ki @ Kj^T  (B^T = Kj row-major)
    v8f s_acc[4];
    for (int t = 0; t < 4; ++t)
      for (int e = 0; e < 8; ++e) s_acc[t][e] = 0.f;
    for (int t = 0; t < 4; ++t)
      for (int ks = 0; ks < 2; ++ks) {
        v16h bfr = ld_frag_b(&Kn[(t * 16) * LDT], LDT, lane, ks * 32);
        s_acc[t] = wmma_f16(aKi[ks], bfr, s_acc[t]);
      }

    // online softmax: row = r + 8*sel; 16-lane butterflies give row stats
    for (int r = 0; r < 8; ++r) {
      float rmax = -3.0e38f;
      for (int t = 0; t < 4; ++t) {
        float v = keep[r] != 0.f ? s_acc[t][r] * sm_scale : -1.0e9f;
        s_acc[t][r] = v;
        rmax = fmaxf(rmax, v);
      }
      for (int msk = 1; msk < 16; msk <<= 1)
        rmax = fmaxf(rmax, __shfl_xor(rmax, msk, 32));
      float mnew = fmaxf(m_st[r], rmax);
      float scl = __expf(m_st[r] - mnew);
      m_st[r] = mnew;
      float rsum = 0.f;
      for (int t = 0; t < 4; ++t) {
        float pv = __expf(s_acc[t][r] - mnew);
        s_acc[t][r] = pv;
        rsum += pv;
      }
      for (int msk = 1; msk < 16; msk <<= 1)
        rsum += __shfl_xor(rsum, msk, 32);
      l_st[r] = l_st[r] * scl + rsum;
      for (int t = 0; t < 4; ++t) o_acc[t][r] *= scl;
      for (int t = 0; t < 4; ++t)  // stage P in f16 (row-major) for A-layout reload
        Pb[wid][(sel * 8 + r) * LDT + t * 16 + nlo] = f2h(s_acc[t][r]);
    }

    // O += P @ Vj   (contraction over j; B^T = Vt rows = d)
    v16h aP[2];
    aP[0] = ld_frag_a(&Pb[wid][0], LDT, lane, 0);
    aP[1] = ld_frag_a(&Pb[wid][0], LDT, lane, 32);
    for (int t = 0; t < 4; ++t)
      for (int ks = 0; ks < 2; ++ks) {
        v16h bfr = ld_frag_b(&Vt[(t * 16) * LDT], LDT, lane, ks * 32);
        o_acc[t] = wmma_f16(aP[ks], bfr, o_acc[t]);
      }
    __syncthreads();
  }

  for (int t = 0; t < 4; ++t)
    for (int r = 0; r < 8; ++r) {
      float v = o_acc[t][r] / l_st[r];
      int row = i0 + wid * 16 + sel * 8 + r;
      int col = h * kHD + t * 16 + nlo;
      wV[(size_t)(b * kSeq + row) * kEmbed + col] = f2h(v);
    }
}

extern "C" void kernel_launch(void* const* d_in, const int* in_sizes, int n_in,
                              void* d_out, int out_size, void* d_ws, size_t ws_size,
                              hipStream_t stream) {
  (void)in_sizes; (void)n_in; (void)out_size; (void)ws_size;
  // setup_inputs order: x, attention_mask, Wq, bq, Wk, bk, Wv, bv, Wo, bo
  const float* x     = (const float*)d_in[0];
  const int*   amask = (const int*)d_in[1];
  const float* Wk    = (const float*)d_in[4];
  const float* bk    = (const float*)d_in[5];
  const float* Wo    = (const float*)d_in[8];
  const float* bo    = (const float*)d_in[9];
  float* out = (float*)d_out;
  // Q (Wq,bq) and Wv,bv are dead in the reference (scores use K both sides, V=K).

  char* ws = (char*)d_ws;
  size_t off = 0;
  auto alloc = [&](size_t bytes) -> void* {
    void* p = ws + off;
    off += (bytes + 255) & ~size_t(255);
    return p;
  };
  const size_t actN = (size_t)kB * kSeq * kEmbed;   // 6,291,456
  const size_t wN   = (size_t)kEmbed * kEmbed;      //   589,824
  unsigned short* xh  = (unsigned short*)alloc(actN * 2);
  unsigned short* Wkh = (unsigned short*)alloc(wN * 2);
  unsigned short* Woh = (unsigned short*)alloc(wN * 2);
  unsigned short* Khb = (unsigned short*)alloc(actN * 2);
  unsigned short* wVh = (unsigned short*)alloc(actN * 2);  // ~40 MB total

  int nx4 = (int)(actN / 4);
  cvt_f32_f16<<<nx4 / 256, 256, 0, stream>>>(x, xh, nx4);
  int nw4 = (int)(wN / 4);
  cvt_f32_f16<<<nw4 / 256, 256, 0, stream>>>(Wk, Wkh, nw4);
  cvt_f32_f16<<<nw4 / 256, 256, 0, stream>>>(Wo, Woh, nw4);

  dim3 gg(kEmbed / 128, (kB * kSeq) / 128);  // (6, 64)
  gemm_nt_f16<<<gg, 256, 0, stream>>>(xh, Wkh, bk, (void*)Khb,
                                      kB * kSeq, kEmbed, kEmbed, 0);

  dim3 fg(kSeq / 64, kB * kH);  // (16, 96)
  flash_attn_kernel<<<fg, 128, 0, stream>>>(Khb, amask, wVh);

  gemm_nt_f16<<<gg, 256, 0, stream>>>(wVh, Woh, bo, (void*)out,
                                      kB * kSeq, kEmbed, kEmbed, 1);
}
